// PartPVTConv_65180423684304
// MI455X (gfx1250) — matrix operand order repaired and checked by reading, not compiled, not run on hardware
//
#include <hip/hip_runtime.h>
#include <hip/hip_bf16.h>

#define B_ 8
#define N_ 4096
#define C_ 64
#define R_ 32
#define R3_ 32768

typedef __attribute__((ext_vector_type(16))) _Float16 v16h;
typedef __attribute__((ext_vector_type(8)))  _Float16 v8h;
typedef __attribute__((ext_vector_type(8)))  float    v8f;

// ---------------- WMMA helpers (gfx1250, wave32) ----------------

__device__ __forceinline__ v8f wmma16(v16h a, v16h b, v8f c) {
    return __builtin_amdgcn_wmma_f32_16x16x32_f16(false, a, false, b,
                                                  (short)0, c, false, false);
}

// A-matrix 16x32 f16 fragment. Lane lrow = row M; half hf selects K subset:
// hf=0 -> K = {0..7} U {16..23}; hf=1 -> K = {8..15} U {24..31}.
__device__ __forceinline__ v16h loadA(const _Float16* base, int row, int ld, int hf) {
    const _Float16* p = base + (size_t)row * ld + hf * 8;
    v8h lo = *(const v8h*)(p);
    v8h hi = *(const v8h*)(p + 16);
    v16h r;
#pragma unroll
    for (int i = 0; i < 8; i++) { r[i] = lo[i]; r[i + 8] = hi[i]; }
    return r;
}

// B-matrix 32x16 f16 fragment. Lane lrow = column N; half hf selects K subset:
// hf=0 -> K = {0..7} U {16..23}; hf=1 -> K = {8..15} U {24..31}. Memory holds B^T row-major, ld in halves.
__device__ __forceinline__ v16h loadB(const _Float16* base, int row, int ld, int hf) {
    const _Float16* p = base + (size_t)row * ld + hf * 8;
    v8h lo = *(const v8h*)(p);
    v8h hi = *(const v8h*)(p + 16);
    v16h r;
#pragma unroll
    for (int i = 0; i < 8; i++) { r[i] = lo[i]; r[i + 8] = hi[i]; }
    return r;
}

__device__ __forceinline__ float redmax16(float v) {
    v = fmaxf(v, __shfl_xor(v, 1));
    v = fmaxf(v, __shfl_xor(v, 2));
    v = fmaxf(v, __shfl_xor(v, 4));
    v = fmaxf(v, __shfl_xor(v, 8));
    return v;
}
__device__ __forceinline__ float redsum16(float v) {
    v += __shfl_xor(v, 1);
    v += __shfl_xor(v, 2);
    v += __shfl_xor(v, 4);
    v += __shfl_xor(v, 8);
    return v;
}

// async global->LDS copy of one 4KB tile (32 keys x 64 ch f16): 8 x b128/lane.
// GVS addressing: mem = SGPR_base + VGPR_offset. Tracked on ASYNCcnt.
__device__ __forceinline__ void stage_tile_async(const _Float16* gbase, _Float16* lds,
                                                 int lane) {
    unsigned lb = (unsigned)(size_t)lds + (unsigned)lane * 16u;  // flat[31:0] == LDS offset
    unsigned gb = (unsigned)lane * 16u;
#pragma unroll
    for (int i = 0; i < 8; i++) {
        asm volatile("global_load_async_to_lds_b128 %0, %1, %2 offset:0"
                     :: "v"(lb + (unsigned)i * 512u),
                        "v"(gb + (unsigned)i * 512u),
                        "s"(gbase)
                     : "memory");
    }
}

// ---------------- glue kernels ----------------

__global__ void k_zero(float* p, long long n) {
    long long i = (long long)blockIdx.x * 256 + threadIdx.x;
    if (i < n) p[i] = 0.0f;
}

// pack conv weights to [tap][o][i] f16, point weights to f16, features to [b][n][c] f16
__global__ void k_pack(const float* c1w, const float* c2w,
                       const float* wq, const float* wk, const float* wv, const float* wo,
                       const float* feats,
                       _Float16* w1h, _Float16* w2h,
                       _Float16* wqh, _Float16* wkh, _Float16* wvh, _Float16* woh,
                       _Float16* fh) {
    int i = blockIdx.x * 256 + threadIdx.x;
    if (i < 27 * 64 * 64) {
        int t = i / 4096, r = i % 4096, o = r / 64, c = r % 64;
        w1h[i] = (_Float16)c1w[(o * 64 + c) * 27 + t];
        w2h[i] = (_Float16)c2w[(o * 64 + c) * 27 + t];
    }
    if (i < 4096) {
        wqh[i] = (_Float16)wq[i]; wkh[i] = (_Float16)wk[i];
        wvh[i] = (_Float16)wv[i]; woh[i] = (_Float16)wo[i];
    }
    if (i < B_ * N_ * C_) {
        int b = i / (N_ * C_); int r = i % (N_ * C_); int n = r / C_; int c = r % C_;
        fh[i] = (_Float16)feats[(size_t)b * C_ * N_ + (size_t)c * N_ + n];
    }
}

// per-batch coord mean and max centered norm
__global__ void k_stats(const float* coords, float* stats) {
    int b = blockIdx.x, tid = threadIdx.x;
    __shared__ float red[256];
    const float* cb = coords + (size_t)b * 3 * N_;
    float s0 = 0, s1 = 0, s2 = 0;
    for (int n = tid; n < N_; n += 256) { s0 += cb[n]; s1 += cb[N_ + n]; s2 += cb[2 * N_ + n]; }
    float mean[3]; float sv[3] = { s0, s1, s2 };
    for (int d = 0; d < 3; d++) {
        red[tid] = sv[d]; __syncthreads();
        for (int o = 128; o > 0; o >>= 1) { if (tid < o) red[tid] += red[tid + o]; __syncthreads(); }
        mean[d] = red[0] * (1.0f / N_); __syncthreads();
    }
    float mx = 0.0f;
    for (int n = tid; n < N_; n += 256) {
        float dx = cb[n] - mean[0], dy = cb[N_ + n] - mean[1], dz = cb[2 * N_ + n] - mean[2];
        mx = fmaxf(mx, sqrtf(dx * dx + dy * dy + dz * dz));
    }
    red[tid] = mx; __syncthreads();
    for (int o = 128; o > 0; o >>= 1) { if (tid < o) red[tid] = fmaxf(red[tid], red[tid + o]); __syncthreads(); }
    if (tid == 0) {
        stats[b * 4 + 0] = mean[0]; stats[b * 4 + 1] = mean[1]; stats[b * 4 + 2] = mean[2];
        stats[b * 4 + 3] = 1.0f / (2.0f * red[0]);
    }
}

// scatter-average voxelization (sums + counts via float atomics) + nc store
__global__ void k_vox(const float* coords, const float* feats, const float* stats,
                      float* ncbuf, float* gsum, float* gcnt) {
    int i = blockIdx.x * 256 + threadIdx.x;       // exactly B*N threads
    int b = i / N_, n = i % N_;
    const float* cb = coords + (size_t)b * 3 * N_;
    float inv = stats[b * 4 + 3];
    int vi[3];
#pragma unroll
    for (int d = 0; d < 3; d++) {
        float c = (cb[(size_t)d * N_ + n] - stats[b * 4 + d]) * inv + 0.5f;
        float nc = fminf(fmaxf(c * (float)R_, 0.0f), (float)R_ - 1.0f);
        ncbuf[(size_t)b * 3 * N_ + (size_t)d * N_ + n] = nc;
        vi[d] = (int)roundf(nc);
    }
    int idx = (vi[0] * R_ + vi[1]) * R_ + vi[2];
    float* gs = gsum + ((size_t)b * R3_ + idx) * C_;
    const float* fb = feats + (size_t)b * C_ * N_ + n;
#pragma unroll 8
    for (int c = 0; c < C_; c++) atomicAdd(&gs[c], fb[(size_t)c * N_]);
    atomicAdd(&gcnt[b * R3_ + idx], 1.0f);
}

// grid = sum / max(cnt,1), converted to f16, layout (B, R3, C)
__global__ void k_finalize(const float* gsum, const float* gcnt, _Float16* xh) {
    size_t i = (size_t)blockIdx.x * 256 + threadIdx.x;  // exactly B*R3*C threads
    size_t bv = i >> 6;
    float cnt = fmaxf(gcnt[bv], 1.0f);
    xh[i] = (_Float16)(gsum[i] / cnt);
}

// ---------------- implicit-GEMM 3x3x3 conv via WMMA ----------------
// one wave: 16 out-channels x 16 voxels; K-loop = 27 taps x 2 slices of 32
// (fully unrolled -> dx/dy/dz are immediates, 54 static WMMAs)
__global__ void __launch_bounds__(256) k_conv(const _Float16* __restrict__ xin,
                                              const _Float16* __restrict__ wh,
                                              const float* __restrict__ bias,
                                              _Float16* __restrict__ outh,
                                              float* __restrict__ outf) {
    int wid  = blockIdx.x * 8 + (threadIdx.x >> 5);
    int lane = threadIdx.x & 31, hf = lane >> 4, lrow = lane & 15;
    int ot = wid & 3;
    int vt = (wid >> 2) & 2047;
    int b  = wid >> 13;
    int vcol = vt * 16 + lrow;
    int x = vcol >> 10, y = (vcol >> 5) & 31, z = vcol & 31;
    const _Float16* xb = xin + (size_t)b * R3_ * C_;
    v8f acc = {};
#pragma unroll
    for (int t = 0; t < 27; t++) {
        const int dx = t / 9 - 1, dy = (t / 3) % 3 - 1, dz = t % 3 - 1;
        int nx = x + dx, ny = y + dy, nz = z + dz;
        bool ok = ((unsigned)nx < 32u) && ((unsigned)ny < 32u) && ((unsigned)nz < 32u);
        int nidx = ok ? ((nx * R_ + ny) * R_ + nz) : 0;
        const _Float16* col = xb + (size_t)nidx * C_;
        const _Float16* wt  = wh + t * 4096 + ot * 1024;
#pragma unroll
        for (int kt = 0; kt < 2; kt++) {
            v16h a = loadA(wt + kt * 32, lrow, 64, hf);
            v16h bf = {};
            if (ok) bf = loadB(col + kt * 32, 0, 0, hf);
            acc = wmma16(a, bf, acc);
        }
    }
    size_t obase = ((size_t)b * R3_ + vcol) * C_ + ot * 16 + hf * 8;
    if (outh) {
        v8h oh;
#pragma unroll
        for (int r = 0; r < 8; r++)
            oh[r] = (_Float16)fmaxf(acc[r] + bias[ot * 16 + hf * 8 + r], 0.0f);
        *(v8h*)(outh + obase) = oh;
    }
    if (outf) {
#pragma unroll
        for (int r = 0; r < 8; r++)
            outf[obase + r] = fmaxf(acc[r] + bias[ot * 16 + hf * 8 + r], 0.0f);
    }
}

// SE: global mean over voxels, 64->8 relu, 8->64 sigmoid
__global__ void k_se(const float* h2, const float* w1, const float* b1,
                     const float* w2, const float* b2, float* sescale) {
    int b = blockIdx.x, tid = threadIdx.x, c = tid & 63, part = tid >> 6;
    __shared__ float acc[4][64];
    __shared__ float z[64];
    __shared__ float z1[8];
    const float* base = h2 + (size_t)b * R3_ * C_;
    float s = 0.0f;
    for (int v = part; v < R3_; v += 4) s += base[(size_t)v * C_ + c];
    acc[part][c] = s; __syncthreads();
    if (part == 0) z[c] = (acc[0][c] + acc[1][c] + acc[2][c] + acc[3][c]) * (1.0f / R3_);
    __syncthreads();
    if (tid < 8) {
        float t = b1[tid];
        for (int i = 0; i < 64; i++) t += z[i] * w1[i * 8 + tid];
        z1[tid] = fmaxf(t, 0.0f);
    }
    __syncthreads();
    if (tid < 64) {
        float t = b2[tid];
        for (int j = 0; j < 8; j++) t += z1[j] * w2[j * 64 + tid];
        sescale[b * 64 + tid] = 1.0f / (1.0f + expf(-t));
    }
}

// trilinear devoxelize (SE scale folded in); writes vox branch into d_out
__global__ void k_devox(const float* h2f, const float* ncbuf, const float* sescale,
                        float* out) {
    int tid = threadIdx.x, c = tid & 63;
    int p = blockIdx.x * 4 + (tid >> 6);
    int b = p >> 12, n = p & (N_ - 1);
    const float* nb = ncbuf + (size_t)b * 3 * N_;
    float ncx = nb[n], ncy = nb[N_ + n], ncz = nb[2 * N_ + n];
    float fx = fminf(fmaxf(floorf(ncx), 0.0f), 31.0f);
    float fy = fminf(fmaxf(floorf(ncy), 0.0f), 31.0f);
    float fz = fminf(fmaxf(floorf(ncz), 0.0f), 31.0f);
    int i0x = (int)fx, i0y = (int)fy, i0z = (int)fz;
    int i1x = min(i0x + 1, 31), i1y = min(i0y + 1, 31), i1z = min(i0z + 1, 31);
    float dx = ncx - fx, dy = ncy - fy, dz = ncz - fz;
    const float* hb = h2f + (size_t)b * R3_ * C_;
    float a = 0.0f;
#pragma unroll
    for (int k = 0; k < 8; k++) {
        int xi = (k & 4) ? i1x : i0x, yi = (k & 2) ? i1y : i0y, zi = (k & 1) ? i1z : i0z;
        float w = ((k & 4) ? dx : 1.0f - dx) * ((k & 2) ? dy : 1.0f - dy) * ((k & 1) ? dz : 1.0f - dz);
        a += w * hb[((size_t)((xi * R_ + yi) * R_ + zi)) * C_ + c];
    }
    out[((size_t)b * C_ + c) * N_ + n] = a * sescale[b * 64 + c];
}

// ---------------- attention: QKV projection (WMMA) ----------------
__global__ void __launch_bounds__(256) k_qkv(const _Float16* __restrict__ fh,
                                             const _Float16* __restrict__ wqh,
                                             const _Float16* __restrict__ wkh,
                                             const _Float16* __restrict__ wvh,
                                             _Float16* __restrict__ qh,
                                             _Float16* __restrict__ kh,
                                             _Float16* __restrict__ vt) {
    int wid  = blockIdx.x * 8 + (threadIdx.x >> 5);
    int lane = threadIdx.x & 31, hf = lane >> 4, lrow = lane & 15;
    int ot = wid & 3, nt = (wid >> 2) & 255, b = wid >> 10;
    const _Float16* fb = fh + (size_t)b * N_ * C_ + nt * 1024;
    v8f aq = {}, ak = {}, av = {};
#pragma unroll
    for (int kt = 0; kt < 2; kt++) {
        v16h bcol = loadB(fb + kt * 32, lrow, 64, hf);
        aq = wmma16(loadA(wqh + ot * 1024 + kt * 32, lrow, 64, hf), bcol, aq);
        ak = wmma16(loadA(wkh + ot * 1024 + kt * 32, lrow, 64, hf), bcol, ak);
        av = wmma16(loadA(wvh + ot * 1024 + kt * 32, lrow, 64, hf), bcol, av);
    }
    int n = nt * 16 + lrow;
    v8h qv, kv;
#pragma unroll
    for (int r = 0; r < 8; r++) {
        qv[r] = (_Float16)(aq[r] * 0.125f);   // fold 1/sqrt(64)
        kv[r] = (_Float16)ak[r];
        int d = ot * 16 + hf * 8 + r;
        vt[(size_t)b * C_ * N_ + (size_t)d * N_ + n] = (_Float16)av[r];  // V^T
    }
    *(v8h*)(qh + ((size_t)b * N_ + n) * C_ + ot * 16 + hf * 8) = qv;
    *(v8h*)(kh + ((size_t)b * N_ + n) * C_ + ot * 16 + hf * 8) = kv;
}

// ---------------- flash attention core ----------------
// 4 waves/block; K tiles (32 keys x 64 ch f16 = 4KB) double-buffered in LDS via
// GLOBAL_LOAD_ASYNC_TO_LDS_B128, overlapped with WMMA on the previous tile.
__global__ void __launch_bounds__(128) k_flash(const _Float16* __restrict__ qh,
                                               const _Float16* __restrict__ kh,
                                               const _Float16* __restrict__ vt,
                                               _Float16* __restrict__ pth) {
    __shared__ __align__(32) _Float16 ldsK[4][2][32 * 64];  // 32KB
    __shared__ __align__(32) _Float16 ldsP[4][16 * 40];     // 5KB
    int w = threadIdx.x >> 5;
    int lane = threadIdx.x & 31, hf = lane >> 4, lrow = lane & 15;
    int wid = blockIdx.x * 4 + w;
    int nt = wid & 255, b = wid >> 8;
    const _Float16* qb = qh + (size_t)b * N_ * C_ + nt * 1024;
    v16h aq0 = loadA(qb, lrow, 64, hf);
    v16h aq1 = loadA(qb + 32, lrow, 64, hf);
    v8f O[4] = {};
    float M[8], L[8];
#pragma unroll
    for (int r = 0; r < 8; r++) { M[r] = -1e30f; L[r] = 0.0f; }
    _Float16* P = &ldsP[w][0];
    const _Float16* kbase = kh + (size_t)b * N_ * C_;
    const _Float16* vbase = vt + (size_t)b * C_ * N_;

    // prologue: stage first K tile
    stage_tile_async(kbase, &ldsK[w][0][0], lane);

    for (int mp = 0; mp < N_ / 32; mp++) {
        int mbase = mp * 32;
        const _Float16* lk = &ldsK[w][mp & 1][0];
        if (mp + 1 < N_ / 32) {
            // prefetch next K tile into the other buffer while current computes
            stage_tile_async(kbase + (size_t)(mp + 1) * 32 * C_,
                             &ldsK[w][(mp + 1) & 1][0], lane);
            asm volatile("s_wait_asynccnt 0x8" ::: "memory");  // current tile done
        } else {
            asm volatile("s_wait_asynccnt 0x0" ::: "memory");
        }
        // scores for two 16-key tiles from LDS (K = 64 channels -> 2 WMMAs each)
        v8f sA = {}, sB = {};
        sA = wmma16(aq0, loadB(lk, lrow, 64, hf), sA);
        sA = wmma16(aq1, loadB(lk + 32, lrow, 64, hf), sA);
        const _Float16* lk1 = lk + 16 * 64;
        sB = wmma16(aq0, loadB(lk1, lrow, 64, hf), sB);
        sB = wmma16(aq1, loadB(lk1 + 32, lrow, 64, hf), sB);
        // online softmax per query row (rows r+8*hf live across 16 lanes)
#pragma unroll
        for (int r = 0; r < 8; r++) {
            float rm = redmax16(fmaxf(sA[r], sB[r]));
            float nm = fmaxf(M[r], rm);
            float corr = expf(M[r] - nm);
            float pa = expf(sA[r] - nm);
            float pb = expf(sB[r] - nm);
            float rs = redsum16(pa + pb);
            L[r] = L[r] * corr + rs;
            M[r] = nm;
#pragma unroll
            for (int t = 0; t < 4; t++) O[t][r] *= corr;
            int row = hf * 8 + r;
            P[row * 40 + lrow]      = (_Float16)pa;
            P[row * 40 + 16 + lrow] = (_Float16)pb;
        }
        asm volatile("s_wait_dscnt 0x0" ::: "memory");   // wave-local LDS RAW fence
        v16h ap = loadA(P, lrow, 40, hf);                // P tile 16x32 as A fragment
#pragma unroll
        for (int t = 0; t < 4; t++)
            O[t] = wmma16(ap, loadB(vbase + (size_t)(t * 16) * N_ + mbase, lrow, N_, hf), O[t]);
    }
    // write pt (f16, layout [b][n][d]) normalized by row sums
#pragma unroll
    for (int t = 0; t < 4; t++) {
#pragma unroll
        for (int r = 0; r < 8; r++) {
            int n = nt * 16 + hf * 8 + r;
            pth[((size_t)b * N_ + n) * C_ + t * 16 + lrow] = (_Float16)(O[t][r] / L[r]);
        }
    }
}

// output projection + ReLU, accumulated onto devox result in d_out
__global__ void __launch_bounds__(256) k_oproj(const _Float16* __restrict__ pth,
                                               const _Float16* __restrict__ woh,
                                               float* __restrict__ out) {
    int wid  = blockIdx.x * 8 + (threadIdx.x >> 5);
    int lane = threadIdx.x & 31, hf = lane >> 4, lrow = lane & 15;
    int ot = wid & 3, nt = (wid >> 2) & 255, b = wid >> 10;
    const _Float16* pb = pth + (size_t)b * N_ * C_ + nt * 1024;
    v8f acc = {};
#pragma unroll
    for (int kt = 0; kt < 2; kt++)
        acc = wmma16(loadA(woh + ot * 1024 + kt * 32, lrow, 64, hf),
                     loadB(pb + kt * 32, lrow, 64, hf), acc);
    int n = nt * 16 + lrow;
#pragma unroll
    for (int r = 0; r < 8; r++) {
        int o = ot * 16 + hf * 8 + r;
        size_t idx = ((size_t)b * C_ + o) * N_ + n;
        out[idx] += fmaxf(acc[r], 0.0f);
    }
}

__global__ void k_copy(const float* src, float* dst, int n) {
    int i = blockIdx.x * 256 + threadIdx.x;
    if (i < n) dst[i] = src[i];
}

// ---------------- host side ----------------

extern "C" void kernel_launch(void* const* d_in, const int* in_sizes, int n_in,
                              void* d_out, int out_size, void* d_ws, size_t ws_size,
                              hipStream_t stream) {
    (void)in_sizes; (void)n_in; (void)out_size; (void)ws_size;
    const float* features = (const float*)d_in[0];
    const float* coords   = (const float*)d_in[1];
    const float* conv1_w  = (const float*)d_in[2];
    const float* conv1_b  = (const float*)d_in[3];
    const float* conv2_w  = (const float*)d_in[4];
    const float* conv2_b  = (const float*)d_in[5];
    const float* se_w1    = (const float*)d_in[6];
    const float* se_b1    = (const float*)d_in[7];
    const float* se_w2    = (const float*)d_in[8];
    const float* se_b2    = (const float*)d_in[9];
    const float* wq       = (const float*)d_in[10];
    const float* wk       = (const float*)d_in[11];
    const float* wv       = (const float*)d_in[12];
    const float* wo       = (const float*)d_in[13];
    float* out = (float*)d_out;

    char* ws = (char*)d_ws;
    size_t off = 0;
    auto carve = [&](size_t bytes) -> char* {
        char* p = ws + off;
        off += (bytes + 255) & ~(size_t)255;
        return p;
    };
    float*     stats   = (float*)carve(B_ * 4 * sizeof(float));
    _Float16*  w1h     = (_Float16*)carve(27 * 64 * 64 * 2);
    _Float16*  w2h     = (_Float16*)carve(27 * 64 * 64 * 2);
    _Float16*  wqh     = (_Float16*)carve(64 * 64 * 2);
    _Float16*  wkh     = (_Float16*)carve(64 * 64 * 2);
    _Float16*  wvh     = (_Float16*)carve(64 * 64 * 2);
    _Float16*  woh     = (_Float16*)carve(64 * 64 * 2);
    _Float16*  fh      = (_Float16*)carve((size_t)B_ * N_ * C_ * 2);
    float*     ncbuf   = (float*)carve((size_t)B_ * 3 * N_ * 4);
    float*     sescale = (float*)carve(B_ * 64 * 4);
    _Float16*  qh      = (_Float16*)carve((size_t)B_ * N_ * C_ * 2);
    _Float16*  kh      = (_Float16*)carve((size_t)B_ * N_ * C_ * 2);
    _Float16*  vtb     = (_Float16*)carve((size_t)B_ * N_ * C_ * 2);
    _Float16*  pth     = (_Float16*)carve((size_t)B_ * N_ * C_ * 2);
    _Float16*  xh      = (_Float16*)carve((size_t)B_ * R3_ * C_ * 2);
    _Float16*  h1h     = (_Float16*)carve((size_t)B_ * R3_ * C_ * 2);
    float*     gsum    = (float*)carve((size_t)B_ * R3_ * C_ * 4);   // reused as h2f
    float*     gcnt    = (float*)carve((size_t)B_ * R3_ * 4);
    float*     h2f     = gsum;   // conv2 output reuses scatter buffer (gsum dead by then)

    // 1) zero scatter buffers (gsum + gcnt are contiguous)
    long long nz = (long long)B_ * R3_ * C_ + (long long)B_ * R3_;
    k_zero<<<(unsigned)((nz + 255) / 256), 256, 0, stream>>>(gsum, nz);
    // 2) pack weights + transpose features to f16
    k_pack<<<(B_ * N_ * C_) / 256, 256, 0, stream>>>(conv1_w, conv2_w, wq, wk, wv, wo,
                                                     features, w1h, w2h, wqh, wkh, wvh, woh, fh);
    // 3) coord stats, 4) voxelize scatter, 5) normalize grid
    k_stats<<<B_, 256, 0, stream>>>(coords, stats);
    k_vox<<<(B_ * N_) / 256, 256, 0, stream>>>(coords, features, stats, ncbuf, gsum, gcnt);
    k_finalize<<<(unsigned)(((size_t)B_ * R3_ * C_) / 256), 256, 0, stream>>>(gsum, gcnt, xh);
    // 6) conv1 (f16 out) and conv2 (f32 out) via WMMA implicit GEMM
    k_conv<<<8192, 256, 0, stream>>>(xh, w1h, conv1_b, h1h, (float*)nullptr);
    k_conv<<<8192, 256, 0, stream>>>(h1h, w2h, conv2_b, (_Float16*)nullptr, h2f);
    // 7) SE, 8) trilinear devoxelize -> d_out (vox branch)
    k_se<<<B_, 256, 0, stream>>>(h2f, se_w1, se_b1, se_w2, se_b2, sescale);
    k_devox<<<(B_ * N_) / 4, 256, 0, stream>>>(h2f, ncbuf, sescale, out);
    // 9) attention: QKV, async-pipelined flash core, output projection
    k_qkv<<<1024, 256, 0, stream>>>(fh, wqh, wkh, wvh, qh, kh, vtb);
    k_flash<<<512, 128, 0, stream>>>(qh, kh, vtb, pth);
    k_oproj<<<1024, 256, 0, stream>>>(pth, woh, out);
    // 10) coords passthrough (second tuple output)
    k_copy<<<(B_ * 3 * N_) / 256, 256, 0, stream>>>(coords, out + (size_t)B_ * C_ * N_,
                                                    B_ * 3 * N_);
}
